// scoring_model_68015102100257
// MI455X (gfx1250) — compile-verified, matrix operand order checked by reading, not run on hardware
//
#include <hip/hip_runtime.h>
#include <math.h>

// ---------------------------------------------------------------------------
// GVP-GNN for MI455X (gfx1250).  All dense GEMMs via v_wmma_f32_16x16x32_f16
// (f16 operands, f32 accumulate), register-blocked 32x64 per wave (2x4 WMMA
// tiles) so each K-step does 8 WMMAs on 12 b128 loads (1.5 loads/WMMA) and
// the A panel is only swept Nc/64 times.  Vector-channel (3D) math is tiny
// fp32 VALU work, independent of the scalar path; it feeds norm features into
// the zero-padded K-tails of the GEMM A matrices.
// ---------------------------------------------------------------------------

typedef _Float16 f16;
typedef __attribute__((ext_vector_type(16))) _Float16 v16h;
typedef __attribute__((ext_vector_type(8)))  _Float16 v8h;
typedef __attribute__((ext_vector_type(8)))  float    v8f;

#define NN 8192
#define EE 131072
#define EPSV 1e-8f

__device__ __forceinline__ float sigmoidf_(float x) { return 1.0f / (1.0f + __expf(-x)); }

__device__ __forceinline__ v16h ld_frag(const f16* p) {
  v8h lo = *(const v8h*)p;
  v8h hi = *(const v8h*)(p + 16);
  return __builtin_shufflevector(lo, hi, 0,1,2,3,4,5,6,7,8,9,10,11,12,13,14,15);
}

// ---------------------------------------------------------------------------
// Pack fp32 (rows x cols) -> f16 (rows x dstride), zero padding cols..dstride.
// ---------------------------------------------------------------------------
__global__ __launch_bounds__(256) void k_pack_f16(f16* __restrict__ dst,
                                                  const float* __restrict__ src,
                                                  int rows, int cols, int dstride) {
  size_t i = (size_t)blockIdx.x * blockDim.x + threadIdx.x;
  size_t tot = (size_t)rows * (size_t)dstride;
  if (i >= tot) return;
  int r = (int)(i / dstride);
  int c = (int)(i - (size_t)r * dstride);
  dst[i] = (c < cols) ? (f16)src[(size_t)r * cols + c] : (f16)0.0f;
}

__global__ __launch_bounds__(256) void k_zero_f32(float* __restrict__ p, size_t n) {
  size_t i = (size_t)blockIdx.x * blockDim.x + threadIdx.x;
  if (i < n) p[i] = 0.0f;
}

// ---------------------------------------------------------------------------
// WMMA GEMM: C[M,Nc] = A[M,K] * B[Nc,K]^T (+bias[n]) (+ReLU)
// A,B row-major f16, K multiple of 32, M multiple of 32, Nc multiple of 64.
// One wave computes a 32x64 C block = 2x4 WMMA tiles.
// Fragment layout per CDNA5 ISA 7.12.2: lane<16 row=lane holds K {k..k+7,
// k+16..k+23}, lane>=16 row=lane-16 holds K {k+8..k+15, k+24..k+31}.
// B fragment = rows of W (columns of B_math) with identical packing.
// C: vgpr r -> row +r (lanes<16) / +8+r (lanes>=16), lane&15 -> column.
// ---------------------------------------------------------------------------
__global__ __launch_bounds__(256)
void k_gemm_wmma(const f16* __restrict__ A, int lda,
                 const f16* __restrict__ B, int ldb,
                 const float* __restrict__ bias,
                 float* __restrict__ C32, int ldc32,
                 f16* __restrict__ C16, int ldc16,
                 int M, int Nc, int K, int relu) {
  int lane = threadIdx.x & 31;
  int wave = threadIdx.x >> 5;
  int tile = blockIdx.x * 8 + wave;
  int tn = Nc >> 6;                       // 64-wide column blocks
  int tile_m = tile / tn;
  int tile_n = tile - tile_m * tn;
  if (tile_m * 32 >= M) return;           // wave-uniform: EXEC stays all-ones for WMMA
  int m0 = tile_m << 5, n0 = tile_n << 6;
  int row  = lane & 15;
  int koff = (lane >> 4) << 3;            // 0 or 8
  const f16* ap0 = A + (size_t)(m0 + row) * lda + koff;
  const f16* ap1 = ap0 + (size_t)16 * lda;
  const f16* bp0 = B + (size_t)(n0 + row) * ldb + koff;

  v8f acc[2][4];
#pragma unroll
  for (int i = 0; i < 2; ++i)
#pragma unroll
    for (int j = 0; j < 4; ++j) acc[i][j] = {};

  for (int k = 0; k < K; k += 32) {
    v16h a0 = ld_frag(ap0 + k);
    v16h a1 = ld_frag(ap1 + k);
    v16h b[4];
#pragma unroll
    for (int j = 0; j < 4; ++j) b[j] = ld_frag(bp0 + (size_t)(j * 16) * ldb + k);
#pragma unroll
    for (int j = 0; j < 4; ++j) {
      acc[0][j] = __builtin_amdgcn_wmma_f32_16x16x32_f16(false, a0, false, b[j],
                                                         (short)0, acc[0][j], false, false);
      acc[1][j] = __builtin_amdgcn_wmma_f32_16x16x32_f16(false, a1, false, b[j],
                                                         (short)0, acc[1][j], false, false);
    }
  }

#pragma unroll
  for (int j = 0; j < 4; ++j) {
    int n = n0 + j * 16 + row;
    float bv = bias ? bias[n] : 0.0f;
#pragma unroll
    for (int i = 0; i < 2; ++i) {
      int rbase = m0 + i * 16 + ((lane >> 4) << 3);
#pragma unroll
      for (int r = 0; r < 8; ++r) {
        float x = acc[i][j][r] + bv;
        if (relu) x = fmaxf(x, 0.0f);
        if (C32) C32[(size_t)(rbase + r) * ldc32 + n] = x;
        if (C16) C16[(size_t)(rbase + r) * ldc16 + n] = (f16)x;
      }
    }
  }
}

// ---------------------------------------------------------------------------
// Per-edge vector-channel chain for msg0->msg1->msg2 (independent of scalars).
// Writes vn tails (+zero pads) of A0 (stride 288), A1/A2 (stride 160) and the
// final edge vector message emv (E x 2 x 3).
// ---------------------------------------------------------------------------
__global__ __launch_bounds__(256)
void k_edge_vec(const int* __restrict__ ei, const float* __restrict__ v,
                const float* __restrict__ ev,
                const float* __restrict__ wh0, const float* __restrict__ wv0,   // 5x5, 2x5
                const float* __restrict__ wh1, const float* __restrict__ wv1,   // 2x2, 2x2
                const float* __restrict__ wh2, const float* __restrict__ wv2,   // 2x2, 2x2
                f16* __restrict__ A0, f16* __restrict__ A1, f16* __restrict__ A2,
                float* __restrict__ emv, int E) {
  int e = blockIdx.x * blockDim.x + threadIdx.x;
  if (e >= E) return;
  int src = ei[e], dst = ei[E + e];
  float mv[5][3];
  for (int c = 0; c < 3; ++c) {
    mv[0][c] = v[(size_t)src * 6 + c];
    mv[1][c] = v[(size_t)src * 6 + 3 + c];
    mv[2][c] = ev[(size_t)e * 3 + c];
    mv[3][c] = v[(size_t)dst * 6 + c];
    mv[4][c] = v[(size_t)dst * 6 + 3 + c];
  }
  // ---- msg0 (acts=True) ----
  float vh0[5][3];
  f16* a0 = A0 + (size_t)e * 288;
  for (int h = 0; h < 5; ++h) {
    float ns = 0.0f;
    for (int c = 0; c < 3; ++c) {
      float acc = 0.0f;
      for (int i = 0; i < 5; ++i) acc += wh0[h * 5 + i] * mv[i][c];
      vh0[h][c] = acc; ns += acc * acc;
    }
    a0[264 + h] = (f16)sqrtf(fmaxf(ns, EPSV));
  }
  for (int c = 269; c < 288; ++c) a0[c] = (f16)0.0f;
  float v1[2][3];
  for (int o = 0; o < 2; ++o) {
    float ns = 0.0f;
    for (int c = 0; c < 3; ++c) {
      float acc = 0.0f;
      for (int h = 0; h < 5; ++h) acc += wv0[o * 5 + h] * vh0[h][c];
      v1[o][c] = acc; ns += acc * acc;
    }
    float g = sigmoidf_(sqrtf(fmaxf(ns, EPSV)));
    for (int c = 0; c < 3; ++c) v1[o][c] *= g;
  }
  // ---- msg1 (acts=True) ----
  float vh1[2][3];
  f16* a1 = A1 + (size_t)e * 160;
  for (int h = 0; h < 2; ++h) {
    float ns = 0.0f;
    for (int c = 0; c < 3; ++c) {
      float acc = wh1[h * 2 + 0] * v1[0][c] + wh1[h * 2 + 1] * v1[1][c];
      vh1[h][c] = acc; ns += acc * acc;
    }
    a1[128 + h] = (f16)sqrtf(fmaxf(ns, EPSV));
  }
  for (int c = 130; c < 160; ++c) a1[c] = (f16)0.0f;
  float v2[2][3];
  for (int o = 0; o < 2; ++o) {
    float ns = 0.0f;
    for (int c = 0; c < 3; ++c) {
      float acc = wv1[o * 2 + 0] * vh1[0][c] + wv1[o * 2 + 1] * vh1[1][c];
      v2[o][c] = acc; ns += acc * acc;
    }
    float g = sigmoidf_(sqrtf(fmaxf(ns, EPSV)));
    for (int c = 0; c < 3; ++c) v2[o][c] *= g;
  }
  // ---- msg2 (acts=False) ----
  float vh2[2][3];
  f16* a2 = A2 + (size_t)e * 160;
  for (int h = 0; h < 2; ++h) {
    float ns = 0.0f;
    for (int c = 0; c < 3; ++c) {
      float acc = wh2[h * 2 + 0] * v2[0][c] + wh2[h * 2 + 1] * v2[1][c];
      vh2[h][c] = acc; ns += acc * acc;
    }
    a2[128 + h] = (f16)sqrtf(fmaxf(ns, EPSV));
  }
  for (int c = 130; c < 160; ++c) a2[c] = (f16)0.0f;
  for (int o = 0; o < 2; ++o)
    for (int c = 0; c < 3; ++c)
      emv[(size_t)e * 6 + o * 3 + c] = wv2[o * 2 + 0] * vh2[0][c] + wv2[o * 2 + 1] * vh2[1][c];
}

// Gather [s_src | e_s | s_dst] (f16) into A0 cols 0..263.  One wave per edge.
__global__ __launch_bounds__(256)
void k_build_a0(const int* __restrict__ ei, const float* __restrict__ s,
                const float* __restrict__ es, f16* __restrict__ A0, int E) {
  int t = blockIdx.x * blockDim.x + threadIdx.x;
  int e = t >> 5, lane = t & 31;
  if (e >= E) return;
  int src = ei[e], dst = ei[E + e];
  f16* a = A0 + (size_t)e * 288;
  for (int c = lane; c < 128; c += 32) {
    a[c]       = (f16)s[(size_t)src * 128 + c];
    a[136 + c] = (f16)s[(size_t)dst * 128 + c];
  }
  if (lane < 8) a[128 + lane] = (f16)es[(size_t)e * 8 + lane];
}

// Mean-scatter: one wave per edge, float atomics into agg buffers.
__global__ __launch_bounds__(256)
void k_scatter(const int* __restrict__ ei, const float* __restrict__ ms,
               const float* __restrict__ emv, float* __restrict__ agg_s,
               float* __restrict__ agg_v, float* __restrict__ cnt, int E) {
  int t = blockIdx.x * blockDim.x + threadIdx.x;
  int e = t >> 5, lane = t & 31;
  if (e >= E) return;
  int dst = ei[E + e];
  for (int c = lane; c < 128; c += 32)
    atomicAdd(&agg_s[(size_t)dst * 128 + c], ms[(size_t)e * 128 + c]);
  if (lane < 6) atomicAdd(&agg_v[(size_t)dst * 6 + lane], emv[(size_t)e * 6 + lane]);
  if (lane == 0) atomicAdd(&cnt[dst], 1.0f);
}

// ---------------------------------------------------------------------------
// Node update 1: residual + mean-agg, LayerNorm(g0,b0), vector renorm, then
// the ff0/ff1 vector chains.  One wave (32 lanes) per node; 128-ch reductions
// via __shfl_xor (wave32).  Writes s_ln (f32), Af0 scalars+vn tail (f16),
// v_ln, Af1 vn tail, and the ff vector output fv.
// ---------------------------------------------------------------------------
__global__ __launch_bounds__(256)
void k_node_upd1(const float* __restrict__ s_in, const float* __restrict__ v_in,
                 const float* __restrict__ agg_s, const float* __restrict__ agg_v,
                 const float* __restrict__ cnt,
                 const float* __restrict__ g0, const float* __restrict__ b0,
                 const float* __restrict__ whf0, const float* __restrict__ wvf0, // 4x2, 4x4
                 const float* __restrict__ whf1, const float* __restrict__ wvf1, // 4x4, 2x4
                 float* __restrict__ s_ln, float* __restrict__ v_ln,
                 float* __restrict__ fv, f16* __restrict__ Af0, f16* __restrict__ Af1,
                 int n_nodes) {
  int t = blockIdx.x * blockDim.x + threadIdx.x;
  int n = t >> 5, lane = t & 31;
  if (n >= n_nodes) return;
  float inv = 1.0f / fmaxf(cnt[n], 1.0f);
  float x[4], sum = 0.0f;
  for (int j = 0; j < 4; ++j) {
    int c = lane + 32 * j;
    x[j] = s_in[(size_t)n * 128 + c] + agg_s[(size_t)n * 128 + c] * inv;
    sum += x[j];
  }
  for (int off = 16; off > 0; off >>= 1) sum += __shfl_xor(sum, off, 32);
  float mu = sum * (1.0f / 128.0f);
  float vs = 0.0f;
  for (int j = 0; j < 4; ++j) { float d = x[j] - mu; vs += d * d; }
  for (int off = 16; off > 0; off >>= 1) vs += __shfl_xor(vs, off, 32);
  float rstd = rsqrtf(vs * (1.0f / 128.0f) + 1e-5f);
  for (int j = 0; j < 4; ++j) {
    int c = lane + 32 * j;
    float y = (x[j] - mu) * rstd * g0[c] + b0[c];
    s_ln[(size_t)n * 128 + c] = y;
    Af0[(size_t)n * 160 + c] = (f16)y;
  }
  if (lane == 0) {
    float vl[2][3], ns[2];
    for (int i = 0; i < 2; ++i) {
      ns[i] = 0.0f;
      for (int c = 0; c < 3; ++c) {
        float q = v_in[(size_t)n * 6 + i * 3 + c] + agg_v[(size_t)n * 6 + i * 3 + c] * inv;
        vl[i][c] = q; ns[i] += q * q;
      }
      ns[i] = fmaxf(ns[i], EPSV);
    }
    float denom = 1.0f / sqrtf((ns[0] + ns[1]) * 0.5f);
    for (int i = 0; i < 2; ++i)
      for (int c = 0; c < 3; ++c) {
        vl[i][c] *= denom;
        v_ln[(size_t)n * 6 + i * 3 + c] = vl[i][c];
      }
    // ff0 vector chain (acts=True)
    float vhf[4][3];
    for (int h = 0; h < 4; ++h) {
      float q = 0.0f;
      for (int c = 0; c < 3; ++c) {
        float acc = whf0[h * 2 + 0] * vl[0][c] + whf0[h * 2 + 1] * vl[1][c];
        vhf[h][c] = acc; q += acc * acc;
      }
      Af0[(size_t)n * 160 + 128 + h] = (f16)sqrtf(fmaxf(q, EPSV));
    }
    for (int c = 132; c < 160; ++c) Af0[(size_t)n * 160 + c] = (f16)0.0f;
    float fv1[4][3];
    for (int o = 0; o < 4; ++o) {
      float q = 0.0f;
      for (int c = 0; c < 3; ++c) {
        float acc = 0.0f;
        for (int h = 0; h < 4; ++h) acc += wvf0[o * 4 + h] * vhf[h][c];
        fv1[o][c] = acc; q += acc * acc;
      }
      float g = sigmoidf_(sqrtf(fmaxf(q, EPSV)));
      for (int c = 0; c < 3; ++c) fv1[o][c] *= g;
    }
    // ff1 vector chain (acts=False)
    float vh1[4][3];
    for (int h = 0; h < 4; ++h) {
      float q = 0.0f;
      for (int c = 0; c < 3; ++c) {
        float acc = 0.0f;
        for (int i = 0; i < 4; ++i) acc += whf1[h * 4 + i] * fv1[i][c];
        vh1[h][c] = acc; q += acc * acc;
      }
      Af1[(size_t)n * 544 + 512 + h] = (f16)sqrtf(fmaxf(q, EPSV));
    }
    for (int c = 516; c < 544; ++c) Af1[(size_t)n * 544 + c] = (f16)0.0f;
    for (int o = 0; o < 2; ++o)
      for (int c = 0; c < 3; ++c) {
        float acc = 0.0f;
        for (int h = 0; h < 4; ++h) acc += wvf1[o * 4 + h] * vh1[h][c];
        fv[(size_t)n * 6 + o * 3 + c] = acc;
      }
  }
}

// Node update 2: second LayerNorm + outer 0.5*(out + layer_input) residual.
__global__ __launch_bounds__(256)
void k_node_upd2(const float* __restrict__ s_ln, const float* __restrict__ fs,
                 const float* __restrict__ v_ln, const float* __restrict__ fv,
                 const float* __restrict__ g1, const float* __restrict__ b1,
                 const float* __restrict__ s_res, const float* __restrict__ v_res,
                 float* __restrict__ s_next, float* __restrict__ v_next, int n_nodes) {
  int t = blockIdx.x * blockDim.x + threadIdx.x;
  int n = t >> 5, lane = t & 31;
  if (n >= n_nodes) return;
  float x[4], sum = 0.0f;
  for (int j = 0; j < 4; ++j) {
    int c = lane + 32 * j;
    x[j] = s_ln[(size_t)n * 128 + c] + fs[(size_t)n * 128 + c];
    sum += x[j];
  }
  for (int off = 16; off > 0; off >>= 1) sum += __shfl_xor(sum, off, 32);
  float mu = sum * (1.0f / 128.0f);
  float vs = 0.0f;
  for (int j = 0; j < 4; ++j) { float d = x[j] - mu; vs += d * d; }
  for (int off = 16; off > 0; off >>= 1) vs += __shfl_xor(vs, off, 32);
  float rstd = rsqrtf(vs * (1.0f / 128.0f) + 1e-5f);
  for (int j = 0; j < 4; ++j) {
    int c = lane + 32 * j;
    float y = (x[j] - mu) * rstd * g1[c] + b1[c];
    s_next[(size_t)n * 128 + c] = 0.5f * (y + s_res[(size_t)n * 128 + c]);
  }
  if (lane == 0) {
    float vt[2][3], ns[2];
    for (int i = 0; i < 2; ++i) {
      ns[i] = 0.0f;
      for (int c = 0; c < 3; ++c) {
        float q = v_ln[(size_t)n * 6 + i * 3 + c] + fv[(size_t)n * 6 + i * 3 + c];
        vt[i][c] = q; ns[i] += q * q;
      }
      ns[i] = fmaxf(ns[i], EPSV);
    }
    float denom = 1.0f / sqrtf((ns[0] + ns[1]) * 0.5f);
    for (int i = 0; i < 2; ++i)
      for (int c = 0; c < 3; ++c) {
        float vo = vt[i][c] * denom;
        v_next[(size_t)n * 6 + i * 3 + c] = 0.5f * (vo + v_res[(size_t)n * 6 + i * 3 + c]);
      }
  }
}

// Final head: sigmoid(s @ w_out^T + b_out).  One wave per node.
__global__ __launch_bounds__(256)
void k_out(const float* __restrict__ s, const float* __restrict__ wo,
           const float* __restrict__ bo, float* __restrict__ out, int n_nodes) {
  int t = blockIdx.x * blockDim.x + threadIdx.x;
  int n = t >> 5, lane = t & 31;
  if (n >= n_nodes) return;
  float p = 0.0f;
  for (int c = lane; c < 128; c += 32) p += s[(size_t)n * 128 + c] * wo[c];
  for (int off = 16; off > 0; off >>= 1) p += __shfl_xor(p, off, 32);
  if (lane == 0) out[n] = sigmoidf_(p + bo[0]);
}

// ---------------------------------------------------------------------------
// Host launcher
// ---------------------------------------------------------------------------
extern "C" void kernel_launch(void* const* d_in, const int* in_sizes, int n_in,
                              void* d_out, int out_size, void* d_ws, size_t ws_size,
                              hipStream_t stream) {
  (void)in_sizes; (void)n_in; (void)out_size; (void)ws_size;
  const float* node_s = (const float*)d_in[0];
  const float* node_v = (const float*)d_in[1];
  const int*   eidx   = (const int*)d_in[2];
  const float* edge_s = (const float*)d_in[3];
  const float* edge_v = (const float*)d_in[4];
  // params: 8 blocks x 24 leaves (jax pytree order: list order, dict keys sorted)
  enum { B0=0, B1=1, FF0_BS=2, FF0_WH=3, FF0_WS=4, FF0_WV=5,
         FF1_BS=6, FF1_WH=7, FF1_WS=8, FF1_WV=9, G0=10, G1=11,
         M0_BS=12, M0_WH=13, M0_WS=14, M0_WV=15,
         M1_BS=16, M1_WH=17, M1_WS=18, M1_WV=19,
         M2_BS=20, M2_WH=21, M2_WS=22, M2_WV=23 };
  auto L = [&](int blk, int leaf) -> const float* {
    return (const float*)d_in[5 + blk * 24 + leaf];
  };
  const float* w_in  = (const float*)d_in[5 + 8 * 24 + 0];
  const float* b_in  = (const float*)d_in[5 + 8 * 24 + 1];
  const float* w_out = (const float*)d_in[5 + 8 * 24 + 2];
  const float* b_out = (const float*)d_in[5 + 8 * 24 + 3];
  float* out = (float*)d_out;

  // workspace bump allocator (256B aligned)
  char* base = (char*)d_ws;
  size_t off = 0;
  auto alloc = [&](size_t bytes) -> void* {
    void* p = base + off;
    off += (bytes + 255) & ~(size_t)255;
    return p;
  };
  f16* A_in = (f16*)alloc((size_t)NN * 1440 * 2);
  f16* Winp = (f16*)alloc((size_t)128 * 1440 * 2);
  f16* W0p  = (f16*)alloc((size_t)8 * 128 * 288 * 2);
  f16* W1p  = (f16*)alloc((size_t)8 * 128 * 160 * 2);
  f16* W2p  = (f16*)alloc((size_t)8 * 128 * 160 * 2);
  f16* Wf0p = (f16*)alloc((size_t)8 * 512 * 160 * 2);
  f16* Wf1p = (f16*)alloc((size_t)8 * 128 * 544 * 2);
  float* sA   = (float*)alloc((size_t)NN * 128 * 4);
  float* sB   = (float*)alloc((size_t)NN * 128 * 4);
  float* vA   = (float*)alloc((size_t)NN * 6 * 4);
  float* vB   = (float*)alloc((size_t)NN * 6 * 4);
  float* s_ln = (float*)alloc((size_t)NN * 128 * 4);
  float* v_ln = (float*)alloc((size_t)NN * 6 * 4);
  float* fvb  = (float*)alloc((size_t)NN * 6 * 4);
  float* fsb  = (float*)alloc((size_t)NN * 128 * 4);
  float* aggblk = (float*)alloc((size_t)NN * 135 * 4);  // agg_s | agg_v | cnt
  float* agg_s = aggblk;
  float* agg_v = aggblk + (size_t)NN * 128;
  float* cntp  = aggblk + (size_t)NN * 134;
  f16* Af0 = (f16*)alloc((size_t)NN * 160 * 2);
  f16* Af1 = (f16*)alloc((size_t)NN * 544 * 2);
  f16* A0  = (f16*)alloc((size_t)EE * 288 * 2);
  f16* A1  = (f16*)alloc((size_t)EE * 160 * 2);
  f16* A2  = (f16*)alloc((size_t)EE * 160 * 2);
  float* ms  = (float*)alloc((size_t)EE * 128 * 4);
  float* emv = (float*)alloc((size_t)EE * 6 * 4);

  auto pack = [&](f16* dst, const float* src, int rows, int cols, int ds) {
    size_t tot = (size_t)rows * ds;
    k_pack_f16<<<dim3((unsigned)((tot + 255) / 256)), dim3(256), 0, stream>>>(dst, src, rows, cols, ds);
  };
  auto gemm = [&](const f16* A, int lda, const f16* B, int ldb, const float* bias,
                  float* C32, int ldc32, f16* C16, int ldc16, int M, int Nc, int K, int relu) {
    int tiles = (M / 32) * (Nc / 64);     // 32x64 C-block per wave
    k_gemm_wmma<<<dim3((unsigned)((tiles + 7) / 8)), dim3(256), 0, stream>>>(
        A, lda, B, ldb, bias, C32, ldc32, C16, ldc16, M, Nc, K, relu);
  };

  // -------- pack activations and all ws weights to padded f16 --------
  pack(A_in, node_s, NN, 1433, 1440);
  pack(Winp, w_in, 128, 1433, 1440);
  for (int l = 0; l < 8; ++l) {
    pack(W0p  + (size_t)l * 128 * 288, L(l, M0_WS),  128, 269, 288);
    pack(W1p  + (size_t)l * 128 * 160, L(l, M1_WS),  128, 130, 160);
    pack(W2p  + (size_t)l * 128 * 160, L(l, M2_WS),  128, 130, 160);
    pack(Wf0p + (size_t)l * 512 * 160, L(l, FF0_WS), 512, 132, 160);
    pack(Wf1p + (size_t)l * 128 * 544, L(l, FF1_WS), 128, 516, 544);
  }

  // -------- input projection: s0 = node_s @ w_in^T + b_in --------
  gemm(A_in, 1440, Winp, 1440, b_in, sA, 128, nullptr, 0, NN, 128, 1440, 0);

  const float* s_cur = sA;
  const float* v_cur = node_v;
  float* sbuf[2] = {sA, sB};
  float* vbuf[2] = {vA, vB};

  for (int l = 0; l < 8; ++l) {
    float* s_nxt = sbuf[(l + 1) & 1];
    float* v_nxt = vbuf[l & 1];

    // zero aggregation block
    {
      size_t n = (size_t)NN * 135;
      k_zero_f32<<<dim3((unsigned)((n + 255) / 256)), dim3(256), 0, stream>>>(aggblk, n);
    }
    // per-edge vector chains (write vn tails + pads + emv)
    k_edge_vec<<<dim3(EE / 256), dim3(256), 0, stream>>>(
        eidx, v_cur, edge_v,
        L(l, M0_WH), L(l, M0_WV), L(l, M1_WH), L(l, M1_WV), L(l, M2_WH), L(l, M2_WV),
        A0, A1, A2, emv, EE);
    // gather scalar message inputs
    k_build_a0<<<dim3(EE * 32 / 256), dim3(256), 0, stream>>>(eidx, s_cur, edge_s, A0, EE);
    // msg GVP scalar GEMM chain
    gemm(A0, 288, W0p + (size_t)l * 128 * 288, 288, L(l, M0_BS),
         nullptr, 0, A1, 160, EE, 128, 288, 1);
    gemm(A1, 160, W1p + (size_t)l * 128 * 160, 160, L(l, M1_BS),
         nullptr, 0, A2, 160, EE, 128, 160, 1);
    gemm(A2, 160, W2p + (size_t)l * 128 * 160, 160, L(l, M2_BS),
         ms, 128, nullptr, 0, EE, 128, 160, 0);
    // mean aggregation
    k_scatter<<<dim3(EE * 32 / 256), dim3(256), 0, stream>>>(eidx, ms, emv, agg_s, agg_v, cntp, EE);
    // node update 1: residual + LN + ff vector chains
    k_node_upd1<<<dim3(NN * 32 / 256), dim3(256), 0, stream>>>(
        s_cur, v_cur, agg_s, agg_v, cntp, L(l, G0), L(l, B0),
        L(l, FF0_WH), L(l, FF0_WV), L(l, FF1_WH), L(l, FF1_WV),
        s_ln, v_ln, fvb, Af0, Af1, NN);
    // ff scalar GEMMs
    gemm(Af0, 160, Wf0p + (size_t)l * 512 * 160, 160, L(l, FF0_BS),
         nullptr, 0, Af1, 544, NN, 512, 160, 1);
    gemm(Af1, 544, Wf1p + (size_t)l * 128 * 544, 544, L(l, FF1_BS),
         fsb, 128, nullptr, 0, NN, 128, 544, 0);
    // node update 2: second LN + outer 0.5 residual
    k_node_upd2<<<dim3(NN * 32 / 256), dim3(256), 0, stream>>>(
        s_ln, fsb, v_ln, fvb, L(l, G1), L(l, B1), s_cur, v_cur, s_nxt, v_nxt, NN);

    s_cur = s_nxt;
    v_cur = v_nxt;
  }

  // -------- output head --------
  k_out<<<dim3(NN * 32 / 256), dim3(256), 0, stream>>>(s_cur, w_out, b_out, out, NN);
}
